// MoTEmbed_27333171872220
// MI455X (gfx1250) — compile-verified, matrix operand order checked by reading, not run on hardware
//
#include <hip/hip_runtime.h>
#include <hip/hip_bf16.h>

typedef __attribute__((ext_vector_type(16))) __bf16 v16bf;
typedef __attribute__((ext_vector_type(8)))  float  v8f;
typedef __attribute__((ext_vector_type(4)))  unsigned int u32x4;
typedef __attribute__((ext_vector_type(8)))  int          i32x8;
typedef __attribute__((ext_vector_type(4)))  int          i32x4;

#define D_DIM   2048
#define NTOK    16384      // B*S = 4*4096
#define BM      128
#define BN      128
#define BK      64
#define LDW     36         // LDS row stride in dwords: 64 bf16 = 32 dw + 4 dw pad (144B, 16B-aligned)
#define THREADS 256

#if defined(__has_builtin)
#  if __has_builtin(__builtin_amdgcn_tensor_load_to_lds)
#    define HAVE_TDM 1
#  endif
#endif
#ifndef HAVE_TDM
#  define HAVE_TDM 0
#endif

__device__ __forceinline__ unsigned int pack_bf16(float lo, float hi) {
  unsigned int a = __float_as_uint(lo);
  unsigned int b = __float_as_uint(hi);
  a = (a + 0x7FFFu + ((a >> 16) & 1u)) >> 16;   // round-to-nearest-even
  b = (b + 0x7FFFu + ((b >> 16) & 1u)) >> 16;
  return a | (b << 16);
}

// Assemble one 16x32 (A) or 32x16 (B) bf16 fragment per the CDNA5 ISA layout:
// lanes 0-15 hold K = {0..7, 16..23}, lanes 16-31 hold K = {8..15, 24..31}.
// Two groups of 4 consecutive dwords, 16B-aligned -> ds_load_b128 pairs.
__device__ __forceinline__ v16bf load_frag(const unsigned int* __restrict__ row,
                                           int kk, int half) {
  union { unsigned int u[8]; v16bf v; } f;
  const int b0 = (kk + half * 8) >> 1;          // dword index of K base
#pragma unroll
  for (int p = 0; p < 4; ++p) {
    f.u[p]     = row[b0 + p];                   // K base + {0,2,4,6}
    f.u[p + 4] = row[b0 + 8 + p];               // K base + 16 + {0,2,4,6}
  }
  return f.v;
}

#if HAVE_TDM
// Issue one Tensor Data Mover load (D# per ISA ch.8). flags = group0 dword0
// (count=1, optionally gather_mode bit31).
__device__ __forceinline__ void tdm_load(unsigned int flags, unsigned int lds_addr,
                                         unsigned long long gaddr,
                                         i32x8 g1, i32x4 g2, i32x4 g3) {
  u32x4 g0;
  g0[0] = flags;
  g0[1] = lds_addr;                                            // LDS byte address
  g0[2] = (unsigned int)(gaddr & 0xFFFFFFFFull);               // global_addr[31:0]
  g0[3] = (unsigned int)((gaddr >> 32) & 0x1FFFFFFull) | (2u << 30); // addr hi, type=2
#if defined(__clang_major__) && __clang_major__ >= 23
  i32x8 g4; g4[0]=0;g4[1]=0;g4[2]=0;g4[3]=0;g4[4]=0;g4[5]=0;g4[6]=0;g4[7]=0;
  __builtin_amdgcn_tensor_load_to_lds(g0, g1, g2, g3, g4, 0);
#else
  __builtin_amdgcn_tensor_load_to_lds(g0, g1, g2, g3, 0);
#endif
}
#endif

__global__ __launch_bounds__(THREADS) void build_lists_kernel(
    const int* __restrict__ type_ids, int* __restrict__ counts,
    int* __restrict__ idx0, int* __restrict__ idx1) {
  int t = blockIdx.x * blockDim.x + threadIdx.x;
  if (t < NTOK) {
    int e = type_ids[t];
    if (e == 0) { int p = atomicAdd(&counts[0], 1); idx0[p] = t; }
    else        { int p = atomicAdd(&counts[1], 1); idx1[p] = t; }
  }
}

// fp32 -> packed bf16 pairs (one dword per pair)
__global__ __launch_bounds__(THREADS) void cvt_f32_bf16_kernel(
    const float* __restrict__ src, unsigned int* __restrict__ dst, int npairs) {
  int i = blockIdx.x * blockDim.x + threadIdx.x;
  int stride = gridDim.x * blockDim.x;
  for (; i < npairs; i += stride) {
    float2 f = ((const float2*)src)[i];
    dst[i] = pack_bf16(f.x, f.y);
  }
}

// ---------------------------------------------------------------------------
// Tier-1 GEMM: X and W pre-converted to bf16 in d_ws. ALL tile movement done
// by the Tensor Data Mover into double-buffered LDS:
//   - B tile: one 2D TDM load (128 W-rows x 64 K), padded to LDW=36.
//   - A tile: 8 gather-mode TDM loads, 16 token-rows each, row indices =
//     compacted token ids packed as 16-bit indices in D# groups 2/3.
// Wave 0 issues 9 DMAs one K-step ahead; s_wait_tensorcnt(9) overlaps the
// in-flight set with compute. The hot loop is pure ds_load_b128 + v_wmma.
// ---------------------------------------------------------------------------
__global__ __launch_bounds__(THREADS) void moe_gemm_tdm_kernel(
    const unsigned int* __restrict__ Xb,   // NTOK x (D_DIM/2) dwords of bf16 pairs
    const unsigned int* __restrict__ Wb,   // 2 x D_DIM x (D_DIM/2) dwords
    const float* __restrict__ b0, const float* __restrict__ b1,
    const int* __restrict__ counts,
    const int* __restrict__ idx0, const int* __restrict__ idx1,
    float* __restrict__ Y)
{
  const int e   = blockIdx.z;
  const int cnt = counts[e];
  const int tm  = blockIdx.y;
  if (tm * BM >= cnt) return;
  const int tn = blockIdx.x;

  const int*   idx  = e ? idx1 : idx0;
  const float* bias = e ? b1 : b0;

  __shared__ unsigned int sA[2][BM * LDW];
  __shared__ unsigned int sB[2][BN * LDW];
  __shared__ int sTok[BM];

  const int tid  = threadIdx.x;
  const int lane = tid & 31;
  const int wid  = tid >> 5;
  const int half = lane >> 4;
  const int l15  = lane & 15;

  if (tid < BM) {
    int gm = tm * BM + tid;
    sTok[tid] = (gm < cnt) ? idx[gm] : -1;
  }
  __syncthreads();

  const int wm = wid >> 1;   // 4 waves along M (32 rows each)
  const int wn = wid & 1;    // 2 waves along N (64 cols each)

  v8f acc[2][4];
#pragma unroll
  for (int mt = 0; mt < 2; ++mt)
#pragma unroll
    for (int nt = 0; nt < 4; ++nt)
#pragma unroll
      for (int j = 0; j < 8; ++j) acc[mt][nt][j] = 0.0f;

#if HAVE_TDM
  // --- descriptor templates (uniform, ISA ch.8 bit layout) ---
  i32x8 g1b;                                 // B tile: normal 2D mode
  g1b[0] = (1 << 16)                         // data_size = 2 bytes
         | (1 << 20)                         // pad_enable
         | (4 << 22)                         // pad_interval: 32 dwords (one row)
         | (3 << 25);                        // pad_amount: 4 dwords -> LDW = 36
  g1b[1] = (int)(0x800u << 16);              // tensor_dim0 = 2048
  g1b[2] = (int)(0x800u << 16);              // tensor_dim1 = 2048
  g1b[3] = (int)(64u << 16);                 // tile_dim0 = 64 (K)
  g1b[4] = 128;                              // tile_dim1 = 128 (rows)
  g1b[5] = 2048;                             // tensor_dim0_stride
  g1b[6] = 0; g1b[7] = 0;

  i32x8 g1a = g1b;                           // A tile: gather mode template
  g1a[2] = (int)(0x4000u << 16);             // tensor_dim1 = 16384 (all tokens)

  i32x4 z4; z4[0] = 0; z4[1] = 0; z4[2] = 0; z4[3] = 0;

  // Pack 16-bit row indices (token ids) for 8 gather descriptors.
  int vcnt[8]; i32x4 gi2[8], gi3[8];
  const int mbase = tm * BM;
#pragma unroll
  for (int d = 0; d < 8; ++d) {
    int rem = cnt - (mbase + d * 16);
    vcnt[d] = rem >= 16 ? 16 : (rem > 0 ? rem : 0);
#pragma unroll
    for (int j = 0; j < 4; ++j) {
      int rA = mbase + d * 16 + 2 * j;       // rows 0..7 of this descriptor
      gi2[d][j] = (idx[rA] & 0xFFFF) | ((idx[rA + 1] & 0xFFFF) << 16);
      int rB = rA + 8;                       // rows 8..15
      gi3[d][j] = (idx[rB] & 0xFFFF) | ((idx[rB + 1] & 0xFFFF) << 16);
    }
  }

  const unsigned long long wBase = (unsigned long long)(size_t)Wb
      + ((unsigned long long)e * D_DIM * D_DIM
         + (unsigned long long)(tn * BN) * D_DIM) * 2ull;
  const unsigned long long xBase = (unsigned long long)(size_t)Xb;

  auto issue_set = [&](int ktI, int p) {
    tdm_load(1u, (unsigned int)(size_t)&sB[p][0],
             wBase + (unsigned long long)ktI * 2ull, g1b, z4, z4);
#pragma unroll
    for (int d = 0; d < 8; ++d) {
      i32x8 g1 = g1a; g1[4] = vcnt[d];       // tile_dim1 = #valid indices
      tdm_load(1u | (1u << 31),              // count=1 | gather_mode (16-bit idx)
               (unsigned int)(size_t)&sA[p][d * 16 * LDW],
               xBase + (unsigned long long)ktI * 2ull, g1, gi2[d], gi3[d]);
    }
  };

  if (wid == 0) issue_set(0, 0);             // prologue: fill buffer 0

  int p = 0;
  for (int kt = 0; kt < D_DIM; kt += BK, p ^= 1) {
    if (wid == 0) {
      if (kt + BK < D_DIM) {
        issue_set(kt + BK, p ^ 1);           // stream next buffer in background
        __builtin_amdgcn_s_wait_tensorcnt(9);// in-order: buffer p's 9 DMAs done
      } else {
        __builtin_amdgcn_s_wait_tensorcnt(0);
      }
    }
    __syncthreads();                         // buffer p visible to all waves

#pragma unroll
    for (int kk = 0; kk < BK; kk += 32) {
      v16bf a[2], b[4];
#pragma unroll
      for (int mt = 0; mt < 2; ++mt)
        a[mt] = load_frag(&sA[p][(wm * 32 + mt * 16 + l15) * LDW], kk, half);
#pragma unroll
      for (int nt = 0; nt < 4; ++nt)
        b[nt] = load_frag(&sB[p][(wn * 64 + nt * 16 + l15) * LDW], kk, half);
#pragma unroll
      for (int mt = 0; mt < 2; ++mt)
#pragma unroll
        for (int nt = 0; nt < 4; ++nt)
          acc[mt][nt] = __builtin_amdgcn_wmma_f32_16x16x32_bf16(
              false, a[mt], false, b[nt], (short)0, acc[mt][nt], false, false);
    }
    __syncthreads();                         // reads of p done; p reusable at kt+2*BK
  }
#else
  // No TDM builtin: stage bf16 tiles with uint4 copies (single buffer).
  const int ldRow = tid >> 1;
  const int ldCol = (tid & 1) * 32;
  const int tokMine = sTok[ldRow];
  const uint4* __restrict__ srcA16 = (tokMine >= 0)
      ? (const uint4*)(Xb + (size_t)tokMine * (D_DIM / 2)) : (const uint4*)0;
  const uint4* __restrict__ srcB16 =
      (const uint4*)(Wb + (size_t)e * D_DIM * (D_DIM / 2)
                        + (size_t)(tn * BN + ldRow) * (D_DIM / 2));
  uint4* __restrict__ dstA4 = (uint4*)&sA[0][ldRow * LDW + (ldCol >> 1)];
  uint4* __restrict__ dstB4 = (uint4*)&sB[0][ldRow * LDW + (ldCol >> 1)];

  for (int kt = 0; kt < D_DIM; kt += BK) {
#pragma unroll
    for (int v = 0; v < 4; ++v) {
      uint4 d = srcA16 ? srcA16[(((kt >> 1) + (ldCol >> 1)) >> 2) + v]
                       : make_uint4(0u, 0u, 0u, 0u);
      dstA4[v] = d;
    }
#pragma unroll
    for (int v = 0; v < 4; ++v)
      dstB4[v] = srcB16[(((kt >> 1) + (ldCol >> 1)) >> 2) + v];
    __syncthreads();
#pragma unroll
    for (int kk = 0; kk < BK; kk += 32) {
      v16bf a[2], b[4];
#pragma unroll
      for (int mt = 0; mt < 2; ++mt)
        a[mt] = load_frag(&sA[0][(wm * 32 + mt * 16 + l15) * LDW], kk, half);
#pragma unroll
      for (int nt = 0; nt < 4; ++nt)
        b[nt] = load_frag(&sB[0][(wn * 64 + nt * 16 + l15) * LDW], kk, half);
#pragma unroll
      for (int mt = 0; mt < 2; ++mt)
#pragma unroll
        for (int nt = 0; nt < 4; ++nt)
          acc[mt][nt] = __builtin_amdgcn_wmma_f32_16x16x32_bf16(
              false, a[mt], false, b[nt], (short)0, acc[mt][nt], false, false);
    }
    __syncthreads();
  }
#endif

  // Epilogue: bias + scatter (C layout: VGPR j -> M = j + 8*half, N = lane&15)
#pragma unroll
  for (int nt = 0; nt < 4; ++nt) {
    const int   ncol = tn * BN + wn * 64 + nt * 16 + l15;
    const float bv   = bias[ncol];
#pragma unroll
    for (int mt = 0; mt < 2; ++mt) {
#pragma unroll
      for (int j = 0; j < 8; ++j) {
        int m   = wm * 32 + mt * 16 + half * 8 + j;
        int tok = sTok[m];
        if (tok >= 0) Y[(long long)tok * D_DIM + ncol] = acc[mt][nt][j] + bv;
      }
    }
  }
}

// ---------------------------------------------------------------------------
// Fallback GEMM: fp32 inputs, convert to bf16 while staging (small d_ws).
// ---------------------------------------------------------------------------
__global__ __launch_bounds__(THREADS) void moe_gemm_fallback_kernel(
    const float* __restrict__ X,
    const float* __restrict__ W0, const float* __restrict__ b0,
    const float* __restrict__ W1, const float* __restrict__ b1,
    const int* __restrict__ counts,
    const int* __restrict__ idx0, const int* __restrict__ idx1,
    float* __restrict__ Y)
{
  const int e   = blockIdx.z;
  const int cnt = counts[e];
  const int tm  = blockIdx.y;
  if (tm * BM >= cnt) return;
  const int tn = blockIdx.x;

  const int*   idx  = e ? idx1 : idx0;
  const float* W    = e ? W1 : W0;
  const float* bias = e ? b1 : b0;

  __shared__ unsigned int sA[BM * LDW];
  __shared__ unsigned int sB[BN * LDW];
  __shared__ int sTok[BM];

  const int tid  = threadIdx.x;
  const int lane = tid & 31;
  const int wid  = tid >> 5;
  const int half = lane >> 4;
  const int l15  = lane & 15;

  if (tid < BM) {
    int gm = tm * BM + tid;
    sTok[tid] = (gm < cnt) ? idx[gm] : -1;
  }
  __syncthreads();

  const int wm = wid >> 1;
  const int wn = wid & 1;

  v8f acc[2][4];
#pragma unroll
  for (int mt = 0; mt < 2; ++mt)
#pragma unroll
    for (int nt = 0; nt < 4; ++nt)
#pragma unroll
      for (int j = 0; j < 8; ++j) acc[mt][nt][j] = 0.0f;

  const int ldRow = tid >> 1;
  const int ldCol = (tid & 1) * 32;
  const int tokMine = sTok[ldRow];
  const float4* __restrict__ srcA4 = (tokMine >= 0)
      ? (const float4*)(X + (long long)tokMine * D_DIM + ldCol) : (const float4*)0;
  const float4* __restrict__ srcB4 =
      (const float4*)(W + (long long)(tn * BN + ldRow) * D_DIM + ldCol);
  unsigned int* __restrict__ dstA = &sA[ldRow * LDW + (ldCol >> 1)];
  unsigned int* __restrict__ dstB = &sB[ldRow * LDW + (ldCol >> 1)];

  for (int kt = 0; kt < D_DIM; kt += BK) {
    if (kt + BK < D_DIM) {
      if (srcA4) __builtin_prefetch((const char*)srcA4 + (size_t)(kt + BK) * 4, 0, 1);
      __builtin_prefetch((const char*)srcB4 + (size_t)(kt + BK) * 4, 0, 1);
    }
#pragma unroll
    for (int v = 0; v < 8; ++v) {
      float4 f = srcA4 ? srcA4[(kt >> 2) + v] : make_float4(0.f, 0.f, 0.f, 0.f);
      dstA[v * 2 + 0] = pack_bf16(f.x, f.y);
      dstA[v * 2 + 1] = pack_bf16(f.z, f.w);
    }
#pragma unroll
    for (int v = 0; v < 8; ++v) {
      float4 f = srcB4[(kt >> 2) + v];
      dstB[v * 2 + 0] = pack_bf16(f.x, f.y);
      dstB[v * 2 + 1] = pack_bf16(f.z, f.w);
    }
    __syncthreads();

#pragma unroll
    for (int kk = 0; kk < BK; kk += 32) {
      v16bf a[2], b[4];
#pragma unroll
      for (int mt = 0; mt < 2; ++mt)
        a[mt] = load_frag(&sA[(wm * 32 + mt * 16 + l15) * LDW], kk, half);
#pragma unroll
      for (int nt = 0; nt < 4; ++nt)
        b[nt] = load_frag(&sB[(wn * 64 + nt * 16 + l15) * LDW], kk, half);
#pragma unroll
      for (int mt = 0; mt < 2; ++mt)
#pragma unroll
        for (int nt = 0; nt < 4; ++nt)
          acc[mt][nt] = __builtin_amdgcn_wmma_f32_16x16x32_bf16(
              false, a[mt], false, b[nt], (short)0, acc[mt][nt], false, false);
    }
    __syncthreads();
  }

#pragma unroll
  for (int nt = 0; nt < 4; ++nt) {
    const int   ncol = tn * BN + wn * 64 + nt * 16 + l15;
    const float bv   = bias[ncol];
#pragma unroll
    for (int mt = 0; mt < 2; ++mt) {
#pragma unroll
      for (int j = 0; j < 8; ++j) {
        int m   = wm * 32 + mt * 16 + half * 8 + j;
        int tok = sTok[m];
        if (tok >= 0) Y[(long long)tok * D_DIM + ncol] = acc[mt][nt][j] + bv;
      }
    }
  }
}

extern "C" void kernel_launch(void* const* d_in, const int* in_sizes, int n_in,
                              void* d_out, int out_size, void* d_ws, size_t ws_size,
                              hipStream_t stream) {
  (void)in_sizes; (void)n_in; (void)out_size;
  const float* X  = (const float*)d_in[0];
  const int*   T  = (const int*)  d_in[1];
  const float* W0 = (const float*)d_in[2];
  const float* b0 = (const float*)d_in[3];
  const float* W1 = (const float*)d_in[4];
  const float* b1 = (const float*)d_in[5];
  float* Y = (float*)d_out;

  // ws layout: [0,256) counters | idx0[NTOK] | idx1[NTOK] | Wb (bf16) | Xb (bf16)
  int* counts = (int*)d_ws;
  int* idx0   = (int*)((char*)d_ws + 256);
  int* idx1   = idx0 + NTOK;

  hipMemsetAsync(counts, 0, 256, stream);
  build_lists_kernel<<<dim3((NTOK + THREADS - 1) / THREADS), dim3(THREADS), 0, stream>>>(
      T, counts, idx0, idx1);

  const size_t off_wb   = 256 + (size_t)2 * NTOK * sizeof(int);          // 256-aligned
  const size_t wb_bytes = (size_t)2 * D_DIM * D_DIM * 2;                 // 16 MB
  const size_t off_xb   = off_wb + wb_bytes;
  const size_t xb_bytes = (size_t)NTOK * D_DIM * 2;                      // 64 MB
  const size_t need     = off_xb + xb_bytes;

  dim3 grid(D_DIM / BN, NTOK / BM, 2);
  if (ws_size >= need) {
    unsigned int* Wb = (unsigned int*)((char*)d_ws + off_wb);
    unsigned int* Xb = (unsigned int*)((char*)d_ws + off_xb);
    const int wpairs = D_DIM * D_DIM / 2;
    cvt_f32_bf16_kernel<<<dim3(2048), dim3(THREADS), 0, stream>>>(W0, Wb, wpairs);
    cvt_f32_bf16_kernel<<<dim3(2048), dim3(THREADS), 0, stream>>>(W1, Wb + wpairs, wpairs);
    cvt_f32_bf16_kernel<<<dim3(4096), dim3(THREADS), 0, stream>>>(X, Xb, NTOK * D_DIM / 2);
    moe_gemm_tdm_kernel<<<grid, dim3(THREADS), 0, stream>>>(
        Xb, Wb, b0, b1, counts, idx0, idx1, Y);
  } else {
    moe_gemm_fallback_kernel<<<grid, dim3(THREADS), 0, stream>>>(
        X, W0, b0, W1, b1, counts, idx0, idx1, Y);
  }
}
